// SpatioTemporalBlock_42099269435631
// MI455X (gfx1250) — matrix operand order repaired and checked by reading, not compiled, not run on hardware
//
#include <hip/hip_runtime.h>
#include <hip/hip_bf16.h>

// ---------------- problem constants ----------------
#define Bb 2
#define Nn 200
#define FM 8
#define HH 128
#define TT 64
#define ER 600
#define EC 6400
#define BT (Bb * TT)      // 128
#define NF (Nn * FM)      // 1600
#define EPSF 1e-5f

typedef __attribute__((ext_vector_type(16))) _Float16 v16h;
typedef __attribute__((ext_vector_type(8)))  float    v8f;

// ---------------- WMMA helpers (gfx1250 wave32, 16x16x32 f16 -> f32) -------
// A (16x32 f16): lane l holds row M=l&15; half=l>>4:
//   element i : K = 16*half + i (contiguous, same as B; kmapA table order unused)
// B (32x16 f16): lane l holds col N=l&15; K = 16*half + i (sequential)
// C/D (16x16 f32): lane l holds col N=l&15; vgpr v holds row M = v + 8*half
__device__ __forceinline__ int kmapA(int i, int half) {
  int j = i >> 1, r = i & 1;
  return (j < 4) ? (half * 8 + 2 * j + r) : (16 + half * 8 + 2 * (j - 4) + r);
}

// 16x64 strip: one A fragment feeds 4 column tiles (A reuse x4).
// arow: lane's A row (f32). brow: lane's W row for column tile 0
// (W[N,K] row-major, lane col = col0+idx16, tile c adds 16*ldb rows).
__device__ __forceinline__ void wmma_strip4(const float* __restrict__ arow,
                                            const float* __restrict__ brow,
                                            int ldb, int K, v8f acc[4], int half) {
  for (int kb = 0; kb < K; kb += 32) {
    v16h a;
#pragma unroll
    for (int i = 0; i < 16; ++i) a[i] = (_Float16)arow[kb + 16 * half + i];
#pragma unroll
    for (int c = 0; c < 4; ++c) {
      const float* br = brow + (size_t)c * 16 * ldb + kb + 16 * half;
      v16h b;
#pragma unroll
      for (int i = 0; i < 16; ++i) b[i] = (_Float16)br[i];   // contiguous -> b128s
      acc[c] = __builtin_amdgcn_wmma_f32_16x16x32_f16(false, a, false, b,
                                                      (short)0, acc[c], false, false);
    }
  }
}

// ---------------- kernel 0: edge weights ----------------
__global__ void edge_w_kernel(const float* __restrict__ attr,
                              const float* __restrict__ Wre,
                              const float* __restrict__ bre,
                              const float* __restrict__ gate_r,
                              const float* __restrict__ cew,
                              const float* __restrict__ gate_c,
                              float* __restrict__ w_r, float* __restrict__ w_c) {
  int i = blockIdx.x * blockDim.x + threadIdx.x;
  float gr = 1.f / (1.f + __expf(-gate_r[0]));
  float gc = 1.f / (1.f + __expf(-gate_c[0]));
  if (i < ER) {
    float s = bre[0];
#pragma unroll
    for (int j = 0; j < 4; ++j) s += attr[i * 4 + j] * Wre[j];
    w_r[i] = fminf(fmaxf(gr * s, 0.f), 1.f);
  }
  if (i < EC) {
    w_c[i] = fminf(fmaxf(gc * cew[i], 0.f), 1.f);
  }
}

// ---------------- kernel 1: x_global [B,N,H,T] -> xg [BT,N,H] ----------------
__global__ void build_xg_kernel(const float* __restrict__ xglob, float* __restrict__ xg) {
  size_t i = (size_t)blockIdx.x * blockDim.x + threadIdx.x;
  if (i >= (size_t)BT * Nn * HH) return;
  int h = (int)(i % HH);
  size_t q = i / HH;
  int n = (int)(q % Nn);
  int bt = (int)(q / Nn);
  int b = bt / TT, t = bt % TT;
  xg[i] = xglob[(((size_t)(b * Nn + n) * HH + h)) * TT + t];
}

// ---------------- kernel 2: per_feature_x [B,N,F,H,T] + emb -> featin [BT,NF,H] ----
__global__ void build_featin_kernel(const float* __restrict__ pfx,
                                    const float* __restrict__ femb,
                                    float* __restrict__ featin) {
  size_t i = (size_t)blockIdx.x * blockDim.x + threadIdx.x;
  if (i >= (size_t)BT * NF * HH) return;
  int h = (int)(i % HH);
  size_t q = i / HH;           // bt*NF + n*FM + f
  int f = (int)(q % FM);
  size_t q2 = q / FM;          // bt*Nn + n
  int n = (int)(q2 % Nn);
  int bt = (int)(q2 / Nn);
  int b = bt / TT, t = bt % TT;
  featin[i] = pfx[(((((size_t)(b * Nn + n)) * FM + f) * HH + h)) * TT + t]
            + femb[f * HH + h];
}

// ---------------- kernel 2b: Wt [H,H,3] -> wt3 [3][Hout][Hin] (contiguous K) ---
__global__ void tr_wt_kernel(const float* __restrict__ Wt, float* __restrict__ wt3) {
  int i = blockIdx.x * blockDim.x + threadIdx.x;   // over 3*H*H
  if (i >= 3 * HH * HH) return;
  int ii = i % HH;
  int q = i / HH;
  int o = q % HH;
  int s = q / HH;
  wt3[i] = Wt[((size_t)o * HH + ii) * 3 + s];
}

// ---------------- kernel 3: temporal conv (pos-enc + Conv1d k=3) via WMMA ----
// block = one (b,n); 256 threads = 8 waves = 4 row-tiles x 2 strips (16x64 each)
__global__ void temporal_conv_kernel(const float* __restrict__ xglob,
                                     const float* __restrict__ pe,
                                     const float* __restrict__ wt3,  // [3][Ho][Hi]
                                     const float* __restrict__ bt_,
                                     float* __restrict__ xtemp) {
  __shared__ float xt_s[(TT + 2) * HH];   // padded [t+1][h], 33792 B
  int bn = blockIdx.x;
  int b = bn / Nn, n = bn % Nn;
  const float* xbase = xglob + (size_t)(b * Nn + n) * HH * TT;
  for (int idx = threadIdx.x; idx < (TT + 2) * HH; idx += blockDim.x) {
    int trow = idx / HH, i = idx % HH;
    int t = trow - 1;
    float v = 0.f;
    if (t >= 0 && t < TT) v = xbase[(size_t)i * TT + t] + pe[t * HH + i];
    xt_s[idx] = v;
  }
  __syncthreads();
  int wave = threadIdx.x >> 5;
  int lane = threadIdx.x & 31;
  int half = lane >> 4, idx16 = lane & 15;
  int rt = wave >> 1;                 // 0..3
  int col0 = (wave & 1) * 64;         // strip
  int t0 = rt * 16;
  float* out = xtemp + (size_t)bn * TT * HH;

  v8f acc[4];
#pragma unroll
  for (int c = 0; c < 4; ++c) {
    float bval = bt_[col0 + 16 * c + idx16];
#pragma unroll
    for (int v = 0; v < 8; ++v) acc[c][v] = bval;
  }
  for (int s = 0; s < 3; ++s) {       // taps: x_t[t + s - 1] -> padded row t0+m+s
    const float* arow = xt_s + (t0 + idx16 + s) * HH;
    const float* brow = wt3 + (size_t)s * HH * HH + (size_t)(col0 + idx16) * HH;
    wmma_strip4(arow, brow, HH, HH, acc, half);
  }
#pragma unroll
  for (int c = 0; c < 4; ++c) {
#pragma unroll
    for (int v = 0; v < 8; ++v) {
      int m = v + 8 * half;
      out[(size_t)(t0 + m) * HH + col0 + 16 * c + idx16] = acc[c][v];
    }
  }
}

// ---------------- kernel 4: generic tiled GEMM  Y = act([X|X2] @ W.T + b) ----
// Wave computes a 16x64 strip. X:[R,K], X2:[R,K] or null,
// W:[Ncols, (X2? 2K : K)] row-major, Y:[R,Ncols]. Ncols multiple of 64.
// act: 0 none, 1 leaky(0.01), 2 silu. store_mode 1: Y[(r/T)*Ncols*T + col*T + r%T]
__global__ void gemm_kernel(const float* __restrict__ X, const float* __restrict__ X2,
                            const float* __restrict__ W, const float* __restrict__ bias,
                            float* __restrict__ Y, int R, int Ncols, int K,
                            int act, int store_mode, int Tdim) {
  int tile = blockIdx.x * (blockDim.x >> 5) + (threadIdx.x >> 5);
  int nst = Ncols >> 6;                      // strips per row tile
  int total = (R >> 4) * nst;
  if (tile >= total) return;                 // wave-uniform: EXEC stays full
  int rt = tile / nst, st = tile % nst;
  int lane = threadIdx.x & 31, half = lane >> 4, idx16 = lane & 15;
  int row0 = rt * 16, col0 = st * 64;
  int ldb = X2 ? 2 * K : K;
  const float* brow = W + (size_t)(col0 + idx16) * ldb;
  v8f acc[4];
#pragma unroll
  for (int c = 0; c < 4; ++c) {
    float bval = bias ? bias[col0 + 16 * c + idx16] : 0.f;
#pragma unroll
    for (int v = 0; v < 8; ++v) acc[c][v] = bval;
  }
  wmma_strip4(X + (size_t)(row0 + idx16) * K, brow, ldb, K, acc, half);
  if (X2)
    wmma_strip4(X2 + (size_t)(row0 + idx16) * K, brow + K, ldb, K, acc, half);
#pragma unroll
  for (int c = 0; c < 4; ++c) {
    int col = col0 + 16 * c + idx16;
#pragma unroll
    for (int v = 0; v < 8; ++v) {
      int m = v + 8 * half;
      float val = acc[c][v];
      if (act == 1)      val = val > 0.f ? val : 0.01f * val;
      else if (act == 2) val = val / (1.f + __expf(-val));
      size_t r = (size_t)row0 + m;
      if (store_mode == 0) {
        Y[r * Ncols + col] = val;
      } else {
        size_t rb = r / Tdim; int rr = (int)(r % Tdim);
        Y[(rb * Ncols + col) * Tdim + rr] = val;
      }
    }
  }
}

// ---------------- kernel 5: scatter-add message passing ----------------
// block = (bt, e); thread = h.  aggr[bt,dst,h] += w[e]*hbuf[bt,src,h]
__global__ void scatter_kernel(const float* __restrict__ hbuf,
                               const int* __restrict__ eidx,
                               const float* __restrict__ w,
                               float* __restrict__ aggr, int E, int nodes) {
  int be = blockIdx.x;
  int bt = be / E, e = be % E;
  int src = eidx[e], dst = eidx[E + e];
  float wv = w[e];
  size_t base = (size_t)bt * nodes * HH;
  float v = wv * hbuf[base + (size_t)src * HH + threadIdx.x];
  atomicAdd(&aggr[base + (size_t)dst * HH + threadIdx.x], v);
}

// ---------------- kernel 6: mean-pool over FM ----------------
__global__ void pool_kernel(const float* __restrict__ fupd, float* __restrict__ pooled) {
  size_t i = (size_t)blockIdx.x * blockDim.x + threadIdx.x;
  if (i >= (size_t)BT * Nn * HH) return;
  int h = (int)(i % HH);
  size_t bn = i / HH;                      // bt*Nn + n
  const float* p = fupd + bn * FM * HH + h;
  float s = 0.f;
#pragma unroll
  for (int f = 0; f < FM; ++f) s += p[(size_t)f * HH];
  pooled[i] = s * (1.f / FM);
}

// ---------------- kernel 7: fused_sp [B,T,N,H] -> xspat [B,N,T,H] ----------------
__global__ void tr_spat_kernel(const float* __restrict__ fused, float* __restrict__ xspat) {
  size_t i = (size_t)blockIdx.x * blockDim.x + threadIdx.x;
  if (i >= (size_t)BT * Nn * HH) return;
  int h = (int)(i % HH);
  size_t q = i / HH;
  int n = (int)(q % Nn);
  int bt = (int)(q / Nn);
  int b = bt / TT, t = bt % TT;
  xspat[(((size_t)(b * Nn + n)) * TT + t) * HH + h] = fused[i];
}

// ---------------- kernel 8: in-place LayerNorm(H) + SiLU, wave per row -------
__global__ void ln_silu_kernel(float* __restrict__ h1, const float* __restrict__ g,
                               const float* __restrict__ be, int rows) {
  int row = blockIdx.x * (blockDim.x >> 5) + (threadIdx.x >> 5);
  if (row >= rows) return;
  int lane = threadIdx.x & 31;
  float* p = h1 + (size_t)row * HH;
  float s = 0.f, s2 = 0.f;
  for (int i = lane; i < HH; i += 32) { float v = p[i]; s += v; s2 += v * v; }
#pragma unroll
  for (int off = 16; off > 0; off >>= 1) {
    s  += __shfl_xor(s,  off, 32);
    s2 += __shfl_xor(s2, off, 32);
  }
  float m = s * (1.f / HH);
  float var = s2 * (1.f / HH) - m * m;
  float rinv = rsqrtf(var + EPSF);
  for (int i = lane; i < HH; i += 32) {
    float v = (p[i] - m) * rinv * g[i] + be[i];
    p[i] = v / (1.f + __expf(-v));
  }
}

// ---------------- kernel 9: per-feature LN + transpose to [B,N,F,H,T] -------
__global__ void pf_ln_out_kernel(const float* __restrict__ fupd,
                                 const float* __restrict__ g,
                                 const float* __restrict__ be,
                                 float* __restrict__ out2) {
  int row = blockIdx.x * (blockDim.x >> 5) + (threadIdx.x >> 5);  // bt*NF + n*FM + f
  if (row >= BT * NF) return;
  int lane = threadIdx.x & 31;
  const float* p = fupd + (size_t)row * HH;
  float s = 0.f, s2 = 0.f;
  for (int i = lane; i < HH; i += 32) { float v = p[i]; s += v; s2 += v * v; }
#pragma unroll
  for (int off = 16; off > 0; off >>= 1) {
    s  += __shfl_xor(s,  off, 32);
    s2 += __shfl_xor(s2, off, 32);
  }
  float m = s * (1.f / HH);
  float var = s2 * (1.f / HH) - m * m;
  float rinv = rsqrtf(var + EPSF);
  int f = row % FM;
  int q = row / FM;                // bt*Nn + n
  int n = q % Nn;
  int bt = q / Nn;
  int b = bt / TT, t = bt % TT;
  size_t obase = ((((size_t)(b * Nn + n)) * FM + f) * HH) * TT + t;
  for (int i = lane; i < HH; i += 32) {
    float v = (p[i] - m) * rinv * g[i] + be[i];
    out2[obase + (size_t)i * TT] = v;
  }
}

// ---------------- host-side orchestration ----------------
extern "C" void kernel_launch(void* const* d_in, const int* in_sizes, int n_in,
                              void* d_out, int out_size, void* d_ws, size_t ws_size,
                              hipStream_t stream) {
  const float* x_global  = (const float*)d_in[0];
  const float* pfx       = (const float*)d_in[1];
  const int*   r_eidx    = (const int*)  d_in[2];
  const float* r_eattr   = (const float*)d_in[3];
  const int*   c_eidx    = (const int*)  d_in[4];
  const float* c_ew      = (const float*)d_in[5];
  const float* pe        = (const float*)d_in[6];
  const float* Wt        = (const float*)d_in[7];
  const float* bt_       = (const float*)d_in[8];
  const float* Wr_lin    = (const float*)d_in[9];
  const float* br_lin    = (const float*)d_in[10];
  const float* Wr_upd    = (const float*)d_in[11];
  const float* br_upd    = (const float*)d_in[12];
  const float* Wr_edge   = (const float*)d_in[13];
  const float* br_edge   = (const float*)d_in[14];
  const float* gate_r    = (const float*)d_in[15];
  const float* Wc_lin    = (const float*)d_in[16];
  const float* bc_lin    = (const float*)d_in[17];
  const float* Wc_upd    = (const float*)d_in[18];
  const float* bc_upd    = (const float*)d_in[19];
  const float* gate_c    = (const float*)d_in[20];
  const float* feat_emb  = (const float*)d_in[21];
  const float* Wsf       = (const float*)d_in[22];
  const float* bsf       = (const float*)d_in[23];
  const float* g_pfn     = (const float*)d_in[24];
  const float* b_pfn     = (const float*)d_in[25];
  const float* Wf1       = (const float*)d_in[26];
  const float* bf1       = (const float*)d_in[27];
  const float* gf_ln     = (const float*)d_in[28];
  const float* bf_ln     = (const float*)d_in[29];
  const float* Wf2       = (const float*)d_in[30];
  const float* bf2       = (const float*)d_in[31];

  float* ws = (float*)d_ws;
  const size_t S_SMALL = (size_t)BT * Nn * HH;   // 3,276,800
  const size_t S_BIG   = (size_t)BT * NF * HH;   // 26,214,400
  float* W_R    = ws + 0;
  float* W_C    = ws + 1024;
  float* WT3    = ws + 8192;                     // 3*128*128 = 49,152
  float* XG     = WT3    + 3 * HH * HH;
  float* HR     = XG     + S_SMALL;
  float* AGGR   = HR     + S_SMALL;
  float* RUPD   = AGGR   + S_SMALL;
  float* XTEMP  = RUPD   + S_SMALL;
  float* FUSED  = XTEMP  + S_SMALL;
  float* FEATIN = FUSED  + S_SMALL;
  float* HC     = FEATIN + S_BIG;
  float* AGGRC  = HC     + S_BIG;
  float* POOLED = AGGR;     // reuse: AGGR dead after river-update GEMM
  float* XSPAT  = HR;       // reuse: HR dead after river scatter
  float* H1     = XG;       // reuse: XG dead after river-update GEMM
  float* FUPD   = HC;       // reuse: HC dead after causal scatter

  float* x_out  = (float*)d_out;                 // [B,N,H,T]
  float* pf_out = (float*)d_out + S_SMALL;       // [B,N,F,H,T]

  // 1) zero scatter targets
  hipMemsetAsync(AGGR,  0, S_SMALL * sizeof(float), stream);
  hipMemsetAsync(AGGRC, 0, S_BIG   * sizeof(float), stream);

  // 2) edge weights + Wt pre-transpose
  edge_w_kernel<<<(EC + 255) / 256, 256, 0, stream>>>(
      r_eattr, Wr_edge, br_edge, gate_r, c_ew, gate_c, W_R, W_C);
  tr_wt_kernel<<<(3 * HH * HH + 255) / 256, 256, 0, stream>>>(Wt, WT3);

  // 3) layouts
  build_xg_kernel<<<(int)((S_SMALL + 255) / 256), 256, 0, stream>>>(x_global, XG);
  build_featin_kernel<<<(int)((S_BIG + 255) / 256), 256, 0, stream>>>(pfx, feat_emb, FEATIN);

  // 4) temporal branch (WMMA, 16x64 strips)
  temporal_conv_kernel<<<Bb * Nn, 256, 0, stream>>>(x_global, pe, WT3, bt_, XTEMP);

  // 5) lin GEMMs (WMMA): HR = XG@Wr_lin.T+b ; HC = FEATIN@Wc_lin.T+b
  {
    int Rr = BT * Nn, Rc = BT * NF;
    gemm_kernel<<<(Rr / 16) * (HH / 64) / 8, 256, 0, stream>>>(
        XG, nullptr, Wr_lin, br_lin, HR, Rr, HH, HH, 0, 0, TT);
    gemm_kernel<<<(Rc / 16) * (HH / 64) / 8, 256, 0, stream>>>(
        FEATIN, nullptr, Wc_lin, bc_lin, HC, Rc, HH, HH, 0, 0, TT);
  }

  // 6) scatter-add message passing
  scatter_kernel<<<BT * ER, HH, 0, stream>>>(HR, r_eidx, W_R, AGGR, ER, Nn);
  scatter_kernel<<<BT * EC, HH, 0, stream>>>(HC, c_eidx, W_C, AGGRC, EC, NF);

  // 7) update GEMMs (concat as two-K accumulation, leaky relu)
  {
    int Rr = BT * Nn, Rc = BT * NF;
    gemm_kernel<<<(Rr / 16) * (HH / 64) / 8, 256, 0, stream>>>(
        AGGR, XG, Wr_upd, br_upd, RUPD, Rr, HH, HH, 1, 0, TT);
    gemm_kernel<<<(Rc / 16) * (HH / 64) / 8, 256, 0, stream>>>(
        AGGRC, FEATIN, Wc_upd, bc_upd, FUPD, Rc, HH, HH, 1, 0, TT);
  }

  // 8) pool over features, spatial fuse (silu), transpose to [B,N,T,H]
  pool_kernel<<<(int)((S_SMALL + 255) / 256), 256, 0, stream>>>(FUPD, POOLED);
  gemm_kernel<<<(BT * Nn / 16) * (HH / 64) / 8, 256, 0, stream>>>(
      RUPD, POOLED, Wsf, bsf, FUSED, BT * Nn, HH, HH, 2, 0, TT);
  tr_spat_kernel<<<(int)((S_SMALL + 255) / 256), 256, 0, stream>>>(FUSED, XSPAT);

  // 9) fusion MLP: H1 = [XTEMP|XSPAT]@Wf1.T+b ; LN+SiLU ; x_out = H1@Wf2.T+b (BNHT store)
  {
    int Rf = Bb * Nn * TT;  // 25600
    gemm_kernel<<<(Rf / 16) * (HH / 64) / 8, 256, 0, stream>>>(
        XTEMP, XSPAT, Wf1, bf1, H1, Rf, HH, HH, 0, 0, TT);
    ln_silu_kernel<<<Rf / 8, 256, 0, stream>>>(H1, gf_ln, bf_ln, Rf);
    gemm_kernel<<<(Rf / 16) * (HH / 64) / 8, 256, 0, stream>>>(
        H1, nullptr, Wf2, bf2, x_out, Rf, HH, HH, 0, 1, TT);
  }

  // 10) per-feature LN + transpose to [B,N,F,H,T]
  pf_ln_out_kernel<<<(BT * NF) / 8, 256, 0, stream>>>(FUPD, g_pfn, b_pfn, pf_out);
}
